// CoaT_27805618275139
// MI455X (gfx1250) — matrix operand order repaired and checked by reading, not compiled, not run on hardware
//
#include <hip/hip_runtime.h>

// Feature selection (probe-confirmed on this toolchain):
//  - __builtin_amdgcn_tensor_load_to_lds: EXISTS, 6-arg form
//  - __builtin_amdgcn_global_load_async_to_lds_b128 / s_wait_asynccnt: EXIST
//  - ds_load_tr16_b128 / global_load_tr16_b128: NOT available
#if __has_builtin(__builtin_amdgcn_tensor_load_to_lds) && __has_builtin(__builtin_amdgcn_s_wait_tensorcnt)
#define USE_TDM 1
#endif
#if __has_builtin(__builtin_amdgcn_global_load_async_to_lds_b128) && __has_builtin(__builtin_amdgcn_s_wait_asynccnt)
#define USE_ASYNC 1
#endif

// ---------------- problem constants ----------------------------------------
constexpr int kB  = 8;
constexpr int kN  = 3137;          // 1 + 56*56
constexpr int kC  = 768;
constexpr int kH  = 8;             // heads
constexpr int kc  = 96;            // head dim
constexpr int kC3 = 3 * kC;        // 2304
constexpr int kM  = kB * kN;       // 25096 rows
constexpr int kHW = 56;

using bf16 = __bf16;
typedef __attribute__((ext_vector_type(16))) __bf16 v16bf;
typedef __attribute__((ext_vector_type(8)))  float  v8f;
typedef unsigned int __attribute__((ext_vector_type(4))) v4u;
typedef int  __attribute__((ext_vector_type(8))) v8i_;
typedef int  __attribute__((ext_vector_type(4))) v4i_;
typedef __attribute__((address_space(3))) char lds_char;
typedef __attribute__((address_space(3))) void lds_void;
typedef __attribute__((address_space(1))) void glob_void;

// ---------------- helpers ---------------------------------------------------
__device__ __forceinline__ v16bf ld16(const bf16* p0, const bf16* p1) {
  union U { v16bf v; uint4 u[2]; } r;
  r.u[0] = *(const uint4*)p0;
  r.u[1] = *(const uint4*)p1;
  return r.v;
}

__device__ __forceinline__ v8f wmma_bf16(v16bf a, v16bf b, v8f c) {
  return __builtin_amdgcn_wmma_f32_16x16x32_bf16(
      false, a, false, b, (short)0, c, false, false);
}

#if defined(USE_TDM)
// Issue one TDM 2D tile load: `rows` x `cols` bf16 tile of a row-major tensor
// with line stride `stride` (elements) from `gsrc` into contiguous LDS at
// `lds_byte_off`.  Must be executed by a single wave; caller waits TENSORcnt.
__device__ __forceinline__ void tdm_load_tile_bf16(const bf16* gsrc,
                                                   unsigned lds_byte_off,
                                                   unsigned cols, unsigned rows,
                                                   unsigned long long stride) {
  unsigned long long ga = (unsigned long long)(size_t)gsrc;
  v4u g0;
  g0[0] = 1u;                                         // count=1, user D#
  g0[1] = lds_byte_off;                               // lds_addr
  g0[2] = (unsigned)ga;                               // global_addr[31:0]
  g0[3] = (unsigned)((ga >> 32) & 0x01FFFFFFu) | (2u << 30);  // addr[56:32], type=2
  unsigned td0 = 0x7FFFFFFFu;                         // huge tensor dims: no clamp
  unsigned td1 = 0x7FFFFFFFu;
  v8i_ g1;
  g1[0] = (int)(1u << 16);                            // data_size=1 -> 2 bytes
  g1[1] = (int)((td0 & 0xFFFFu) << 16);               // tensor_dim0[15:0]
  g1[2] = (int)((td0 >> 16) | ((td1 & 0xFFFFu) << 16));
  g1[3] = (int)((td1 >> 16) | (cols << 16));          // tile_dim0 = cols
  g1[4] = (int)(rows & 0xFFFFu);                      // tile_dim1 = rows, tile_dim2=0
  g1[5] = (int)(unsigned)(stride & 0xFFFFFFFFull);    // tensor_dim0_stride lo
  g1[6] = (int)(unsigned)((stride >> 32) & 0xFFFFull);// stride hi; dim1_stride=0
  g1[7] = 0;
  v4i_ z4 = {};
  v8i_ z8 = {};
  // 6-arg toolchain form: (g0, g1, g2, g3, extra_group, cpol)
  __builtin_amdgcn_tensor_load_to_lds(g0, g1, z4, z4, z8, 0);
}
#endif

// ---------------- K1: f32 -> bf16 convert -----------------------------------
__global__ void cvt_f32_bf16(const float* __restrict__ s, bf16* __restrict__ d, int n) {
  int i = blockIdx.x * blockDim.x + threadIdx.x;
  if (i < n) d[i] = (bf16)s[i];
}

// ---------------- K2: transpose + convert (W[R][K] -> Wt[K][R]) -------------
__global__ void cvt_transpose(const float* __restrict__ s, bf16* __restrict__ d,
                              int R, int K) {
  int i = blockIdx.x * blockDim.x + threadIdx.x;
  if (i < R * K) {
    int r = i / K, k = i - r * K;
    d[(size_t)k * R + r] = (bf16)s[i];
  }
}

// ---------------- K3/K8: tiled bf16 WMMA GEMM -------------------------------
// C[M,Nn] = A[M,K] @ B[K,Nn] + bias.  A,B row-major bf16.
// Block tile 128x64, BK=32, 8 waves in 4x2 grid, 32x32 per wave.
// A tile staged by the Tensor Data Mover (wave 0 issues a 2D D#), falling back
// to async-to-LDS copies, falling back to manual staging.
template <bool OUT_BF16>
__global__ void gemm_bf16_kernel(const bf16* __restrict__ A,
                                 const bf16* __restrict__ Bm,
                                 const float* __restrict__ bias,
                                 void* __restrict__ Out,
                                 int Mdim, int Nn, int Kdim) {
  __shared__ bf16 As[128][32];   // row-major A tile (contiguous 8KB: TDM target)
  __shared__ bf16 Bt[64][32];    // B tile stored transposed: Bt[n][k]

  const int tid  = threadIdx.x;
  const int lane = tid & 31;
  const int wave = tid >> 5;
  const int waveRow = wave >> 1;       // 0..3
  const int waveCol = wave & 1;        // 0..1
  const int rowBase = waveRow * 32;
  const int colBase = waveCol * 32;
  const int n0     = blockIdx.x * 64;
  const int blockM = blockIdx.y * 128;

  v8f acc[2][2] = {};

  for (int k0 = 0; k0 < Kdim; k0 += 32) {
    // ---- stage A tile (128 rows x 32 cols) ----
#if defined(USE_TDM)
    if (tid < 32) {  // wave 0 only: one DMA descriptor for the whole tile
      tdm_load_tile_bf16(A + (size_t)blockM * Kdim + k0,
                         (unsigned)(size_t)(lds_char*)&As[0][0],
                         /*cols=*/32u, /*rows=*/128u,
                         /*stride=*/(unsigned long long)Kdim);
      __builtin_amdgcn_s_wait_tensorcnt(0);
    }
#elif defined(USE_ASYNC)
    {
      int row = tid >> 1, part = tid & 1;          // 2 x 16B per thread
      bf16* gp = const_cast<bf16*>(A) + (size_t)(blockM + row) * Kdim + k0 + part * 16;
      __builtin_amdgcn_global_load_async_to_lds_b128(
          (glob_void*)(void*)gp, (lds_void*)&As[row][part * 16], 0, 0);
      __builtin_amdgcn_global_load_async_to_lds_b128(
          (glob_void*)(void*)(gp + 8), (lds_void*)&As[row][part * 16 + 8], 0, 0);
      __builtin_amdgcn_s_wait_asynccnt(0);
    }
#else
    {
      int row  = tid >> 1;
      int part = tid & 1;
      int gr   = blockM + row;
      uint4 z0{}, z1{};
      if (gr < Mdim) {
        const uint4* gp = (const uint4*)(A + (size_t)gr * Kdim + k0 + part * 16);
        z0 = gp[0];
        z1 = gp[1];
      }
      *(uint4*)&As[row][part * 16]     = z0;
      *(uint4*)&As[row][part * 16 + 8] = z1;
    }
#endif
    // ---- stage B tile transposed (coalesced 16B read, u16 LDS scatter) ----
    {
      int kk = tid >> 3;     // 0..31
      int g  = tid & 7;      // 0..7 -> 8 consecutive columns
      bf16 tmp[8];
      *(uint4*)tmp = *(const uint4*)(Bm + (size_t)(k0 + kk) * Nn + n0 + g * 8);
      if (k0 + 32 < Kdim)
        __builtin_prefetch(Bm + (size_t)(k0 + 32 + kk) * Nn + n0 + g * 8, 0, 1);
#pragma unroll
      for (int i = 0; i < 8; ++i) Bt[g * 8 + i][kk] = tmp[i];
    }
    __syncthreads();

    const int arow = rowBase + (lane & 15);
    const int kb   = (lane >> 4) * 8;    // A K-split per lane-half
    const int kb2  = (lane >> 4) * 16;   // unused: B also uses kb
    const int bcol = colBase + (lane & 15);

    v16bf af0 = ld16(&As[arow][kb],      &As[arow][16 + kb]);
    v16bf af1 = ld16(&As[arow + 16][kb], &As[arow + 16][16 + kb]);
    v16bf bf0 = ld16(&Bt[bcol][kb],      &Bt[bcol][16 + kb]);
    v16bf bf1 = ld16(&Bt[bcol + 16][kb], &Bt[bcol + 16][16 + kb]);

    acc[0][0] = wmma_bf16(af0, bf0, acc[0][0]);
    acc[0][1] = wmma_bf16(af0, bf1, acc[0][1]);
    acc[1][0] = wmma_bf16(af1, bf0, acc[1][0]);
    acc[1][1] = wmma_bf16(af1, bf1, acc[1][1]);
    __syncthreads();
  }

  const int mrow = (lane >> 4) * 8;
#pragma unroll
  for (int mi = 0; mi < 2; ++mi)
#pragma unroll
    for (int ni = 0; ni < 2; ++ni) {
      int col = n0 + colBase + ni * 16 + (lane & 15);
      float bsv = bias[col];
#pragma unroll
      for (int r = 0; r < 8; ++r) {
        int row = blockM + rowBase + mi * 16 + mrow + r;
        if (row < Mdim) {
          float v = acc[mi][ni][r] + bsv;
          if constexpr (OUT_BF16)
            ((bf16*)Out)[(size_t)row * Nn + col] = (bf16)v;
          else
            ((float*)Out)[(size_t)row * Nn + col] = v;
        }
      }
    }
}

// ---------------- K4: softmax stats over tokens for k ------------------------
__global__ void ksoftmax_stats(const bf16* __restrict__ qkv,
                               float* __restrict__ colmax,
                               float* __restrict__ colinv) {
  int bh = blockIdx.x;
  int b = bh >> 3, hd = bh & 7;
  int ch = threadIdx.x;
  if (ch >= kc) return;
  const bf16* base = qkv + (size_t)b * kN * kC3 + kC + hd * kc + ch;
  float mx = -1e30f;
  for (int n = 0; n < kN; ++n)
    mx = fmaxf(mx, (float)base[(size_t)n * kC3]);
  float s = 0.f;
  for (int n = 0; n < kN; ++n)
    s += __expf((float)base[(size_t)n * kC3] - mx);
  colmax[bh * kc + ch] = mx;
  colinv[bh * kc + ch] = 1.0f / s;
}

// ---------------- K5: kv = softmax(k)^T @ v, per (b,head) --------------------
__global__ void kv_kernel(const bf16* __restrict__ qkv,
                          const float* __restrict__ colmax,
                          const float* __restrict__ colinv,
                          bf16* __restrict__ kvb) {
  __shared__ bf16 kt[96][40];  // kt[ch][n_local]
  __shared__ bf16 vt[96][40];  // vt[ch][n_local]
  int bh = blockIdx.x;
  int b = bh >> 3, hd = bh & 7;
  const int tid = threadIdx.x, lane = tid & 31, wave = tid >> 5;
  const bf16* kbase = qkv + (size_t)b * kN * kC3 + kC + hd * kc;
  const bf16* vbase = qkv + (size_t)b * kN * kC3 + 2 * kC + hd * kc;

  v8f acc[5] = {};
  const int nchunks = (kN + 31) / 32;  // 99
  for (int ncu = 0; ncu < nchunks; ++ncu) {
    int nn0 = ncu * 32;
    for (int idx = tid; idx < 32 * kc; idx += 256) {
      int nl = idx / kc, ch = idx - nl * kc;
      int n = nn0 + nl;
      bf16 kvv = (bf16)0.0f, vv = (bf16)0.0f;
      if (n < kN) {
        float kf = (float)kbase[(size_t)n * kC3 + ch];
        kvv = (bf16)(__expf(kf - colmax[bh * kc + ch]) * colinv[bh * kc + ch]);
        vv  = vbase[(size_t)n * kC3 + ch];
      }
      kt[ch][nl] = kvv;
      vt[ch][nl] = vv;
    }
    __syncthreads();
#pragma unroll
    for (int j = 0; j < 5; ++j) {
      int ti = wave + 8 * j;
      if (ti < 36) {                       // wave-uniform branch
        int cc0 = (ti / 6) * 16, dd0 = (ti % 6) * 16;
        int kb  = (lane >> 4) * 8;
        int kb2 = (lane >> 4) * 16;
        int rA = cc0 + (lane & 15);
        int rB = dd0 + (lane & 15);
        v16bf a  = ld16(&kt[rA][kb],  &kt[rA][16 + kb]);
        v16bf bb = ld16(&vt[rB][kb], &vt[rB][16 + kb]);
        acc[j] = wmma_bf16(a, bb, acc[j]);
      }
    }
    __syncthreads();
  }
#pragma unroll
  for (int j = 0; j < 5; ++j) {
    int ti = wave + 8 * j;
    if (ti < 36) {
      int cc0 = (ti / 6) * 16, dd0 = (ti % 6) * 16;
      int dd  = dd0 + (lane & 15);
      int ccb = cc0 + (lane >> 4) * 8;
#pragma unroll
      for (int r = 0; r < 8; ++r)
        kvb[(size_t)bh * (kc * kc) + (ccb + r) * kc + dd] = (bf16)acc[j][r];
    }
  }
}

// ---------------- K6: factor_att = scale * (q @ kv), per (b,head) -----------
__global__ void fa_kernel(const bf16* __restrict__ qkv,
                          const bf16* __restrict__ kvb,
                          bf16* __restrict__ fab) {
  __shared__ bf16 kvT[96][104];  // kvT[dd][cc]
  int bh = blockIdx.y;
  int b = bh >> 3, hd = bh & 7;
  const int tid = threadIdx.x, lane = tid & 31, wave = tid >> 5;

  for (int idx = tid; idx < kc * kc; idx += 256) {
    int cc = idx / kc, dd = idx - cc * kc;
    kvT[dd][cc] = kvb[(size_t)bh * (kc * kc) + idx];
  }
  __syncthreads();

  const int n_base = blockIdx.x * 128 + wave * 16;
  v8f acc[6] = {};
#pragma unroll
  for (int kcn = 0; kcn < 3; ++kcn) {
    int n  = n_base + (lane & 15);
    int kb = (lane >> 4) * 8;
    v16bf a = {};
    if (n < kN) {
      const bf16* p = qkv + (size_t)(b * kN + n) * kC3 + hd * kc + kcn * 32 + kb;
      a = ld16(p, p + 16);
    }
    int kb2 = (lane >> 4) * 16;
#pragma unroll
    for (int t = 0; t < 6; ++t) {
      int dd = t * 16 + (lane & 15);
      v16bf bb = ld16(&kvT[dd][kcn * 32 + kb], &kvT[dd][kcn * 32 + 16 + kb]);
      acc[t] = wmma_bf16(a, bb, acc[t]);
    }
  }
  const float scl = 0.10206207261596577f;  // 96^-0.5
  const int nb = n_base + (lane >> 4) * 8;
#pragma unroll
  for (int t = 0; t < 6; ++t) {
    int dd = t * 16 + (lane & 15);
#pragma unroll
    for (int r = 0; r < 8; ++r) {
      int n = nb + r;
      if (n < kN)
        fab[(size_t)(b * kN + n) * kC + hd * kc + dd] = (bf16)(scl * acc[t][r]);
    }
  }
}

// ---------------- K7: depthwise-conv CRPE epilogue ---------------------------
__global__ void crpe_epilogue(const bf16* __restrict__ qkv,
                              const bf16* __restrict__ fab,
                              const float* __restrict__ c3w, const float* __restrict__ c3b,
                              const float* __restrict__ c5w, const float* __restrict__ c5b,
                              const float* __restrict__ c7w, const float* __restrict__ c7b,
                              bf16* __restrict__ attn) {
  size_t i = (size_t)blockIdx.x * blockDim.x + threadIdx.x;
  if (i >= (size_t)kM * kC) return;
  int ch = (int)(i % kC);
  size_t bn = i / kC;
  int n = (int)(bn % kN);
  int b = (int)(bn / kN);
  float f = (float)fab[i];
  if (n == 0) { attn[i] = (bf16)f; return; }

  int pix = n - 1;
  int y = pix / kHW, x0 = pix - y * kHW;
  int kk; const float* wv; float s;
  if (ch < 2 * kc)      { kk = 3; wv = c3w + ch * 9;              s = c3b[ch]; }
  else if (ch < 5 * kc) { kk = 5; wv = c5w + (ch - 2 * kc) * 25;  s = c5b[ch - 2 * kc]; }
  else                  { kk = 7; wv = c7w + (ch - 5 * kc) * 49;  s = c7b[ch - 5 * kc]; }
  int pp = kk >> 1;
  const bf16* vbase = qkv + (size_t)b * kN * kC3 + 2 * kC + ch;
  for (int ky = 0; ky < kk; ++ky) {
    int yy = y + ky - pp;
    if ((unsigned)yy >= (unsigned)kHW) continue;
    for (int kx = 0; kx < kk; ++kx) {
      int xx = x0 + kx - pp;
      if ((unsigned)xx >= (unsigned)kHW) continue;
      s += wv[ky * kk + kx] * (float)vbase[(size_t)(1 + yy * kHW + xx) * kC3];
    }
  }
  float q = (float)qkv[(size_t)(b * kN + n) * kC3 + ch];
  attn[i] = (bf16)(f + q * s);
}

// ---------------- host side --------------------------------------------------
extern "C" void kernel_launch(void* const* d_in, const int* in_sizes, int n_in,
                              void* d_out, int out_size, void* d_ws, size_t ws_size,
                              hipStream_t stream) {
  const float* x      = (const float*)d_in[0];
  const float* qkv_w  = (const float*)d_in[1];
  const float* qkv_b  = (const float*)d_in[2];
  const float* proj_w = (const float*)d_in[3];
  const float* proj_b = (const float*)d_in[4];
  const float* c3w    = (const float*)d_in[5];
  const float* c3b    = (const float*)d_in[6];
  const float* c5w    = (const float*)d_in[7];
  const float* c5b    = (const float*)d_in[8];
  const float* c7w    = (const float*)d_in[9];
  const float* c7b    = (const float*)d_in[10];
  float* out = (float*)d_out;

  char* ws = (char*)d_ws;
  bf16* xb      = (bf16*)ws; ws += (size_t)kM * kC * 2;        // x bf16, reused as attn
  bf16* wqkvT   = (bf16*)ws; ws += (size_t)kC * kC3 * 2;       // [768][2304]
  bf16* wprojT  = (bf16*)ws; ws += (size_t)kC * kC * 2;        // [768][768]
  bf16* qkv_out = (bf16*)ws; ws += (size_t)kM * kC3 * 2;       // [M][2304]
  float* colmax = (float*)ws; ws += (size_t)kB * kH * kc * 4;
  float* colinv = (float*)ws; ws += (size_t)kB * kH * kc * 4;
  bf16* kvb     = (bf16*)ws; ws += (size_t)kB * kH * kc * kc * 2;
  bf16* fab     = (bf16*)ws; ws += (size_t)kM * kC * 2;        // scale*factor_att
  bf16* attn    = xb;

  { int n = kM * kC;  cvt_f32_bf16<<<(n + 255) / 256, 256, 0, stream>>>(x, xb, n); }
  { int n = kC3 * kC; cvt_transpose<<<(n + 255) / 256, 256, 0, stream>>>(qkv_w, wqkvT, kC3, kC); }
  { int n = kC * kC;  cvt_transpose<<<(n + 255) / 256, 256, 0, stream>>>(proj_w, wprojT, kC, kC); }

  { // qkv GEMM  [M,768] @ [768,2304] + bias -> bf16
    dim3 grid(kC3 / 64, (kM + 127) / 128);
    gemm_bf16_kernel<true><<<grid, 256, 0, stream>>>(xb, wqkvT, qkv_b, qkv_out,
                                                     kM, kC3, kC);
  }

  ksoftmax_stats<<<kB * kH, 128, 0, stream>>>(qkv_out, colmax, colinv);
  kv_kernel<<<kB * kH, 256, 0, stream>>>(qkv_out, colmax, colinv, kvb);

  { dim3 grid((kN + 127) / 128, kB * kH);
    fa_kernel<<<grid, 256, 0, stream>>>(qkv_out, kvb, fab); }

  { size_t total = (size_t)kM * kC;
    crpe_epilogue<<<(unsigned)((total + 255) / 256), 256, 0, stream>>>(
        qkv_out, fab, c3w, c3b, c5w, c5b, c7w, c7b, attn); }

  { // proj GEMM [M,768] @ [768,768] + bias -> fp32 output
    dim3 grid(kC / 64, (kM + 127) / 128);
    gemm_bf16_kernel<false><<<grid, 256, 0, stream>>>(attn, wprojT, proj_b, out,
                                                      kM, kC, kC);
  }
  (void)in_sizes; (void)n_in; (void)out_size; (void)ws_size;
}
